// DVAE_40681930227950
// MI455X (gfx1250) — matrix operand order repaired and checked by reading, not compiled, not run on hardware
//
#include <hip/hip_runtime.h>

// ---------------- types & helpers ----------------
typedef __bf16 bf16_t;
typedef bf16_t v16bf __attribute__((ext_vector_type(16)));
typedef float  v8f   __attribute__((ext_vector_type(8)));
typedef unsigned short u16;

// f32 -> bf16 via hardware convert (RNE)
static __device__ __forceinline__ u16 f2bf(float f) {
  return __builtin_bit_cast(u16, (bf16_t)f);
}

// hardware transcendentals (guarded; fall back to libm if builtin missing)
static __device__ __forceinline__ float fast_tanh(float x) {
#if __has_builtin(__builtin_amdgcn_tanhf)
  return __builtin_amdgcn_tanhf(x);
#else
  return tanhf(x);
#endif
}
static __device__ __forceinline__ float fast_exp2(float x) {
#if __has_builtin(__builtin_amdgcn_exp2f)
  return __builtin_amdgcn_exp2f(x);
#else
  return exp2f(x);
#endif
}
static __device__ __forceinline__ float fast_log2(float x) {
#if __has_builtin(__builtin_amdgcn_logf)
  return __builtin_amdgcn_logf(x);
#else
  return log2f(x);
#endif
}
static __device__ __forceinline__ float fast_rcp(float x) {
#if __has_builtin(__builtin_amdgcn_rcpf)
  return __builtin_amdgcn_rcpf(x);
#else
  return 1.f / x;
#endif
}
#define LOG2E 1.4426950408889634f
#define LN2   0.6931471805599453f
static __device__ __forceinline__ float softplusf(float x) {
  return (x > 20.f) ? x : LN2 * fast_log2(1.f + fast_exp2(x * LOG2E));
}
static __device__ __forceinline__ float sigm(float x) {
  return fast_rcp(1.f + fast_exp2(-x * LOG2E));
}

struct Pack32 { uint4 lo; uint4 hi; };

static __device__ __forceinline__ v16bf frag2(const u16* p0, const u16* p1) {
  Pack32 t;
  t.lo = *reinterpret_cast<const uint4*>(p0);
  t.hi = *reinterpret_cast<const uint4*>(p1);
  return __builtin_bit_cast(v16bf, t);
}

// A fragment (16x32 bf16, row-major source, row stride lda elements)
static __device__ __forceinline__ v16bf load_a(const u16* base, int row0, int lda,
                                               int k0, int lane) {
  int m  = lane & 15;
  int kb = k0 + ((lane & 16) ? 8 : 0);
  const u16* p = base + (row0 + m) * lda + kb;
  return frag2(p, p + 16);              // K = kb..kb+7, kb+16..kb+23
}

// B fragment (32x16): storage is W[n][k] row-major (B[k][n] = W[n][k])
static __device__ __forceinline__ v16bf load_b(const u16* base, int n0, int ldw,
                                               int k0, int lane) {
  int n  = lane & 15;
  int ko = k0 + ((lane & 16) ? 8 : 0);
  const u16* p = base + (n0 + n) * ldw + ko;
  return frag2(p, p + 16);              // K = ko..ko+7, ko+16..ko+23
}

static __device__ __forceinline__ v8f wmma_bf(v16bf a, v16bf b, v8f c) {
  return __builtin_amdgcn_wmma_f32_16x16x32_bf16(false, a, false, b, (short)0, c, false, false);
}

static __device__ __forceinline__ v8f vfill(float v) {
  v8f r;
#pragma unroll
  for (int i = 0; i < 8; ++i) r[i] = v;
  return r;
}

// ---------------- prep kernels ----------------
__global__ void cvt_pad(const float* __restrict__ in, u16* __restrict__ out,
                        int R, int C, int Rp, int Cp) {
  int i = blockIdx.x * blockDim.x + threadIdx.x;
  if (i >= Rp * Cp) return;
  int r = i / Cp, c = i % Cp;
  out[i] = (r < R && c < C) ? f2bf(in[r * C + c]) : (u16)0;
}

// x[B][T][88] f32 -> XBF[T][B][96] bf16 (zero-padded)
__global__ void cvt_x(const float* __restrict__ x, u16* __restrict__ out) {
  int i = blockIdx.x * blockDim.x + threadIdx.x;
  if (i >= 512 * 128 * 96) return;
  int t = i / (128 * 96);
  int r = i % (128 * 96);
  int b = r / 96, d = r % 96;
  out[i] = (d < 88) ? f2bf(x[(b * 512 + t) * 88 + d]) : (u16)0;
}

__global__ void zero_us(u16* __restrict__ p, int n) {
  int i = blockIdx.x * blockDim.x + threadIdx.x;
  if (i < n) p[i] = 0;
}

// ---------------- fused sequential-scan kernel ----------------
// blocks 0..7  : RNN + combiner for batch tile (16 rows), full T loop
// blocks 8..15 : generator transition scan (independent chain)
__global__ __launch_bounds__(256) void dvae_scan(
    const u16* __restrict__ XBF,
    const float* __restrict__ b_ih, const float* __restrict__ b_hh,
    const float* __restrict__ comb_b, const float* __restrict__ mu_b, const float* __restrict__ sig_b,
    const float* __restrict__ g_b1, const float* __restrict__ g_b2,
    const float* __restrict__ h_b1, const float* __restrict__ h_b2,
    const float* __restrict__ mg_b, const float* __restrict__ sg_b,
    const float* __restrict__ eps_comb, const float* __restrict__ eps_init,
    const float* __restrict__ eps_tr,
    const u16* __restrict__ WIH, const u16* __restrict__ WHH, const u16* __restrict__ CW,
    const u16* __restrict__ MUW, const u16* __restrict__ SIW,
    const u16* __restrict__ GW1, const u16* __restrict__ GW2,
    const u16* __restrict__ HW1, const u16* __restrict__ HW2,
    const u16* __restrict__ MGW, const u16* __restrict__ SGW,
    u16* __restrict__ zhat,
    float* __restrict__ mus_inf, float* __restrict__ sig_inf,
    float* __restrict__ mus_gen, float* __restrict__ sig_gen)
{
  __shared__ __align__(16) unsigned char smem[56320];
  const int tid  = threadIdx.x;
  const int wave = __builtin_amdgcn_readfirstlane(tid >> 5);  // provably uniform
  const int lane = tid & 31;
  const int col  = lane & 15;
  const int mOff = (lane & 16) ? 8 : 0;

  if (blockIdx.x < 8) {
    // ---------- forward RNN + combiner ----------
    const int B0 = blockIdx.x * 16;
    float* h_f32 = reinterpret_cast<float*>(smem);            // 16x400 f32
    u16*   h_bf  = reinterpret_cast<u16*>(smem + 25600);      // 16x416 bf16
    u16*   hc_bf = reinterpret_cast<u16*>(smem + 38912);      // 16x416 bf16
    u16*   z_bf  = reinterpret_cast<u16*>(smem + 52224);      // 16x128 bf16

    for (int i = tid; i < 16 * 416; i += 256) { h_bf[i] = 0; hc_bf[i] = 0; }
    for (int i = tid; i < 16 * 128; i += 256) z_bf[i] = 0;
    __syncthreads();

    const int n0 = wave * 16 + col;       // tile nt = wave
    const int n1 = n0 + 128;              // tile nt = wave+8
    const int n2 = n0 + 256;              // tile nt = wave+16
    const int ntail = 384 + col;          // tile 24 (wave 0 only)

    for (int t = 0; t < 512; ++t) {
      // ---- P1: h_new = tanh(x W_ih^T + h W_hh^T + b) : tiles wave,+8,+16 + tail
      const u16* xrow = XBF + t * (128 * 96);
      v8f a0 = vfill(b_ih[n0] + b_hh[n0]);
      v8f a1 = vfill(b_ih[n1] + b_hh[n1]);
      v8f a2 = vfill(b_ih[n2] + b_hh[n2]);
#pragma unroll
      for (int kc = 0; kc < 3; ++kc) {                        // K over D (96)
        v16bf a = load_a(xrow, B0, 96, kc * 32, lane);
        a0 = wmma_bf(a, load_b(WIH, wave * 16,       96, kc * 32, lane), a0);
        a1 = wmma_bf(a, load_b(WIH, wave * 16 + 128, 96, kc * 32, lane), a1);
        a2 = wmma_bf(a, load_b(WIH, wave * 16 + 256, 96, kc * 32, lane), a2);
      }
#pragma unroll
      for (int kc = 0; kc < 13; ++kc) {                       // K over H (416)
        v16bf a = load_a(h_bf, 0, 416, kc * 32, lane);
        a0 = wmma_bf(a, load_b(WHH, wave * 16,       416, kc * 32, lane), a0);
        a1 = wmma_bf(a, load_b(WHH, wave * 16 + 128, 416, kc * 32, lane), a1);
        a2 = wmma_bf(a, load_b(WHH, wave * 16 + 256, 416, kc * 32, lane), a2);
      }
#pragma unroll
      for (int r = 0; r < 8; ++r) {
        a0[r] = fast_tanh(a0[r]); a1[r] = fast_tanh(a1[r]); a2[r] = fast_tanh(a2[r]);
      }
      v8f a3 = vfill(0.f);
      if (wave == 0) {                                        // tail tile 24
        a3 = vfill(b_ih[ntail] + b_hh[ntail]);
#pragma unroll
        for (int kc = 0; kc < 3; ++kc) {
          v16bf a = load_a(xrow, B0, 96, kc * 32, lane);
          a3 = wmma_bf(a, load_b(WIH, 384, 96, kc * 32, lane), a3);
        }
#pragma unroll
        for (int kc = 0; kc < 13; ++kc) {
          v16bf a = load_a(h_bf, 0, 416, kc * 32, lane);
          a3 = wmma_bf(a, load_b(WHH, 384, 416, kc * 32, lane), a3);
        }
#pragma unroll
        for (int r = 0; r < 8; ++r) a3[r] = fast_tanh(a3[r]);
      }
      __syncthreads();
#pragma unroll
      for (int r = 0; r < 8; ++r) {
        int m = r + mOff;
        h_f32[m * 400 + n0] = a0[r];  h_bf[m * 416 + n0] = f2bf(a0[r]);
        h_f32[m * 400 + n1] = a1[r];  h_bf[m * 416 + n1] = f2bf(a1[r]);
        h_f32[m * 400 + n2] = a2[r];  h_bf[m * 416 + n2] = f2bf(a2[r]);
      }
      if (wave == 0) {
#pragma unroll
        for (int r = 0; r < 8; ++r) {
          int m = r + mOff;
          h_f32[m * 400 + ntail] = a3[r];  h_bf[m * 416 + ntail] = f2bf(a3[r]);
        }
      }
      __syncthreads();
      // ---- P2: hc = 0.5*(tanh(z CombW^T + comb_b) + h_t)
      v8f c0 = vfill(comb_b[n0]);
      v8f c1 = vfill(comb_b[n1]);
      v8f c2 = vfill(comb_b[n2]);
#pragma unroll
      for (int kc = 0; kc < 4; ++kc) {                        // K over L (128)
        v16bf a = load_a(z_bf, 0, 128, kc * 32, lane);
        c0 = wmma_bf(a, load_b(CW, wave * 16,       128, kc * 32, lane), c0);
        c1 = wmma_bf(a, load_b(CW, wave * 16 + 128, 128, kc * 32, lane), c1);
        c2 = wmma_bf(a, load_b(CW, wave * 16 + 256, 128, kc * 32, lane), c2);
      }
#pragma unroll
      for (int r = 0; r < 8; ++r) {
        int m = r + mOff;
        hc_bf[m * 416 + n0] = f2bf(0.5f * (fast_tanh(c0[r]) + h_f32[m * 400 + n0]));
        hc_bf[m * 416 + n1] = f2bf(0.5f * (fast_tanh(c1[r]) + h_f32[m * 400 + n1]));
        hc_bf[m * 416 + n2] = f2bf(0.5f * (fast_tanh(c2[r]) + h_f32[m * 400 + n2]));
      }
      if (wave == 0) {                                        // tail tile 24
        v8f c3 = vfill(comb_b[ntail]);
#pragma unroll
        for (int kc = 0; kc < 4; ++kc) {
          v16bf a = load_a(z_bf, 0, 128, kc * 32, lane);
          c3 = wmma_bf(a, load_b(CW, 384, 128, kc * 32, lane), c3);
        }
#pragma unroll
        for (int r = 0; r < 8; ++r) {
          int m = r + mOff;
          hc_bf[m * 416 + ntail] = f2bf(0.5f * (fast_tanh(c3[r]) + h_f32[m * 400 + ntail]));
        }
      }
      __syncthreads();
      // ---- P3: mu / sigma / z sample   (7 L-tiles, one per wave)
      if (wave < 7) {
        int l0 = wave * 16;
        int l  = l0 + col;
        v8f am = vfill((l < 100) ? mu_b[l] : 0.f);
        v8f as = vfill((l < 100) ? sig_b[l] : 0.f);
#pragma unroll
        for (int kc = 0; kc < 13; ++kc) {
          v16bf a = load_a(hc_bf, 0, 416, kc * 32, lane);
          am = wmma_bf(a, load_b(MUW, l0, 416, kc * 32, lane), am);
          as = wmma_bf(a, load_b(SIW, l0, 416, kc * 32, lane), as);
        }
#pragma unroll
        for (int r = 0; r < 8; ++r) {
          if (l < 100) {
            int m   = r + mOff;
            int row = B0 + m;
            int idx = (row * 512 + t) * 100 + l;
            float mu = am[r];
            float sg = softplusf(as[r]);
            mus_inf[idx] = mu;
            sig_inf[idx] = sg;
            float z = mu + sg * eps_comb[idx];
            u16 zb = f2bf(z);
            z_bf[m * 128 + l] = zb;
            zhat[(row * 512 + t) * 128 + l] = zb;
          }
        }
      }
      __syncthreads();
    }
  } else {
    // ---------- generator transition scan ----------
    const int B0 = (blockIdx.x - 8) * 16;
    u16* bufG = reinterpret_cast<u16*>(smem);                 // 16x224 relu(z Gw1)
    u16* bufH = reinterpret_cast<u16*>(smem + 7168);          // 16x224 relu(z Hw1)
    u16* bufP = reinterpret_cast<u16*>(smem + 14336);         // 16x128 relu(hprop)
    u16* z_bf = reinterpret_cast<u16*>(smem + 18432);         // 16x128 z (bf16)

    for (int i = tid; i < 16 * 224; i += 256) { bufG[i] = 0; bufH[i] = 0; }
    for (int i = tid; i < 16 * 128; i += 256) {
      bufP[i] = 0;
      int m = i >> 7, l = i & 127;
      z_bf[i] = (l < 100) ? f2bf(eps_init[(B0 + m) * 100 + l]) : (u16)0;
    }
    for (int i = tid; i < 16 * 100; i += 256) {               // t=0 outputs
      int m = i / 100, l = i % 100;
      int idx = ((B0 + m) * 512) * 100 + l;
      mus_gen[idx] = 0.f;
      sig_gen[idx] = 1.f;
    }
    __syncthreads();

    const int n0  = wave * 16 + col;       // tile nt = wave  (n0 < 128 < 200)
    const int n1  = n0 + 128;              // tile nt = wave+8 (waves 0..4)

    for (int t = 0; t < 511; ++t) {
      // ---- TP1: relu(z Gw1^T + gb1), relu(z Hw1^T + hb1) : tile wave + tail
      v8f g0 = vfill(g_b1[n0]);
      v8f h0 = vfill(h_b1[n0]);
#pragma unroll
      for (int kc = 0; kc < 4; ++kc) {
        v16bf a = load_a(z_bf, 0, 128, kc * 32, lane);
        g0 = wmma_bf(a, load_b(GW1, wave * 16, 128, kc * 32, lane), g0);
        h0 = wmma_bf(a, load_b(HW1, wave * 16, 128, kc * 32, lane), h0);
      }
#pragma unroll
      for (int r = 0; r < 8; ++r) {
        int m = r + mOff;
        bufG[m * 224 + n0] = f2bf(fmaxf(g0[r], 0.f));
        bufH[m * 224 + n0] = f2bf(fmaxf(h0[r], 0.f));
      }
      if (wave < 5) {                                         // tail tiles 8..12
        v8f g1 = vfill((n1 < 200) ? g_b1[n1] : 0.f);
        v8f h1v = vfill((n1 < 200) ? h_b1[n1] : 0.f);
#pragma unroll
        for (int kc = 0; kc < 4; ++kc) {
          v16bf a = load_a(z_bf, 0, 128, kc * 32, lane);
          g1  = wmma_bf(a, load_b(GW1, wave * 16 + 128, 128, kc * 32, lane), g1);
          h1v = wmma_bf(a, load_b(HW1, wave * 16 + 128, 128, kc * 32, lane), h1v);
        }
#pragma unroll
        for (int r = 0; r < 8; ++r) {
          int m = r + mOff;
          bufG[m * 224 + n1] = f2bf(fmaxf(g1[r], 0.f));
          bufH[m * 224 + n1] = f2bf(fmaxf(h1v[r], 0.f));
        }
      }
      __syncthreads();
      // ---- TP2: g = sigmoid(.), hprop, z mgW, mu combine
      v8f muv = vfill(0.f);
      int l = wave * 16 + col;
      if (wave < 7) {
        int l0 = wave * 16;
        v8f ga = vfill((l < 100) ? g_b2[l] : 0.f);
        v8f ha = vfill((l < 100) ? h_b2[l] : 0.f);
        v8f za = vfill((l < 100) ? mg_b[l] : 0.f);
#pragma unroll
        for (int kc = 0; kc < 7; ++kc) {                      // K over HT (224)
          v16bf a1 = load_a(bufG, 0, 224, kc * 32, lane);
          ga = wmma_bf(a1, load_b(GW2, l0, 224, kc * 32, lane), ga);
          v16bf a2 = load_a(bufH, 0, 224, kc * 32, lane);
          ha = wmma_bf(a2, load_b(HW2, l0, 224, kc * 32, lane), ha);
        }
#pragma unroll
        for (int kc = 0; kc < 4; ++kc) {                      // K over L (mg_w)
          v16bf a = load_a(z_bf, 0, 128, kc * 32, lane);
          za = wmma_bf(a, load_b(MGW, l0, 128, kc * 32, lane), za);
        }
#pragma unroll
        for (int r = 0; r < 8; ++r) {
          int m = r + mOff;
          float g  = sigm(ga[r]);
          float hp = ha[r];
          muv[r] = hp * g + za[r] * (1.f - g);
          bufP[m * 128 + l] = f2bf(fmaxf(hp, 0.f));
        }
      }
      __syncthreads();
      // ---- TP3: sigma = softplus(relu(hprop) SgW^T + sg_b), sample z
      if (wave < 7) {
        int l0 = wave * 16;
        v8f sa = vfill((l < 100) ? sg_b[l] : 0.f);
#pragma unroll
        for (int kc = 0; kc < 4; ++kc) {
          v16bf a = load_a(bufP, 0, 128, kc * 32, lane);
          sa = wmma_bf(a, load_b(SGW, l0, 128, kc * 32, lane), sa);
        }
#pragma unroll
        for (int r = 0; r < 8; ++r) {
          if (l < 100) {
            int m   = r + mOff;
            int row = B0 + m;
            float sg = softplusf(sa[r]);
            float z  = muv[r] + sg * eps_tr[(row * 511 + t) * 100 + l];
            int oidx = (row * 512 + (t + 1)) * 100 + l;
            mus_gen[oidx] = muv[r];
            sig_gen[oidx] = sg;
            z_bf[m * 128 + l] = f2bf(z);
          }
        }
      }
      __syncthreads();
    }
  }
}

// ---------------- emission MLP (fully parallel) ----------------
__global__ __launch_bounds__(128) void dvae_emission(
    const u16* __restrict__ zhat,
    const u16* __restrict__ EW1, const u16* __restrict__ EW2, const u16* __restrict__ EW3,
    const float* __restrict__ eb1, const float* __restrict__ eb2, const float* __restrict__ eb3,
    float* __restrict__ xhat)
{
  __shared__ __align__(16) u16 h1[16 * 128];
  __shared__ __align__(16) u16 h2[16 * 128];
  const int tid  = threadIdx.x;
  const int wave = __builtin_amdgcn_readfirstlane(tid >> 5);  // 0..3
  const int lane = tid & 31;
  const int col  = lane & 15;
  const int mOff = (lane & 16) ? 8 : 0;
  const int r0   = blockIdx.x * 16;
  const int n0   = wave * 16 + col;      // tile wave   (n0 < 64)
  const int n1   = n0 + 64;              // tile wave+4 (waves 0..2)

  for (int i = tid; i < 2048; i += 128) { h1[i] = 0; h2[i] = 0; }
  __syncthreads();
  // stage 1: h1 = relu(z EW1^T + eb1)
  {
    v8f a0 = vfill(eb1[n0]);
#pragma unroll
    for (int kc = 0; kc < 4; ++kc) {
      v16bf a = load_a(zhat, r0, 128, kc * 32, lane);
      a0 = wmma_bf(a, load_b(EW1, wave * 16, 128, kc * 32, lane), a0);
    }
#pragma unroll
    for (int r = 0; r < 8; ++r) h1[(r + mOff) * 128 + n0] = f2bf(fmaxf(a0[r], 0.f));
    if (wave < 3) {
      v8f a1 = vfill((n1 < 100) ? eb1[n1] : 0.f);
#pragma unroll
      for (int kc = 0; kc < 4; ++kc) {
        v16bf a = load_a(zhat, r0, 128, kc * 32, lane);
        a1 = wmma_bf(a, load_b(EW1, wave * 16 + 64, 128, kc * 32, lane), a1);
      }
#pragma unroll
      for (int r = 0; r < 8; ++r) h1[(r + mOff) * 128 + n1] = f2bf(fmaxf(a1[r], 0.f));
    }
  }
  __syncthreads();
  // stage 2: h2 = relu(h1 EW2^T + eb2)
  {
    v8f a0 = vfill(eb2[n0]);
#pragma unroll
    for (int kc = 0; kc < 4; ++kc) {
      v16bf a = load_a(h1, 0, 128, kc * 32, lane);
      a0 = wmma_bf(a, load_b(EW2, wave * 16, 128, kc * 32, lane), a0);
    }
#pragma unroll
    for (int r = 0; r < 8; ++r) h2[(r + mOff) * 128 + n0] = f2bf(fmaxf(a0[r], 0.f));
    if (wave < 3) {
      v8f a1 = vfill((n1 < 100) ? eb2[n1] : 0.f);
#pragma unroll
      for (int kc = 0; kc < 4; ++kc) {
        v16bf a = load_a(h1, 0, 128, kc * 32, lane);
        a1 = wmma_bf(a, load_b(EW2, wave * 16 + 64, 128, kc * 32, lane), a1);
      }
#pragma unroll
      for (int r = 0; r < 8; ++r) h2[(r + mOff) * 128 + n1] = f2bf(fmaxf(a1[r], 0.f));
    }
  }
  __syncthreads();
  // stage 3: x_hat = sigmoid(h2 EW3^T + eb3)
  {
    v8f a0 = vfill(eb3[n0]);                                  // n0 < 64 < 88
#pragma unroll
    for (int kc = 0; kc < 4; ++kc) {
      v16bf a = load_a(h2, 0, 128, kc * 32, lane);
      a0 = wmma_bf(a, load_b(EW3, wave * 16, 128, kc * 32, lane), a0);
    }
#pragma unroll
    for (int r = 0; r < 8; ++r) xhat[(r0 + r + mOff) * 88 + n0] = sigm(a0[r]);
    if (wave < 2) {                                           // tiles 4,5 (d 64..95)
      v8f a1 = vfill((n1 < 88) ? eb3[n1] : 0.f);
#pragma unroll
      for (int kc = 0; kc < 4; ++kc) {
        v16bf a = load_a(h2, 0, 128, kc * 32, lane);
        a1 = wmma_bf(a, load_b(EW3, wave * 16 + 64, 128, kc * 32, lane), a1);
      }
#pragma unroll
      for (int r = 0; r < 8; ++r) {
        if (n1 < 88) xhat[(r0 + r + mOff) * 88 + n1] = sigm(a1[r]);
      }
    }
  }
}

// ---------------- host launcher ----------------
extern "C" void kernel_launch(void* const* d_in, const int* in_sizes, int n_in,
                              void* d_out, int out_size, void* d_ws, size_t ws_size,
                              hipStream_t stream) {
  (void)in_sizes; (void)n_in; (void)out_size; (void)ws_size;
  const float* x       = (const float*)d_in[0];
  const float* w_ih    = (const float*)d_in[1];
  const float* w_hh    = (const float*)d_in[2];
  const float* b_ih    = (const float*)d_in[3];
  const float* b_hh    = (const float*)d_in[4];
  const float* comb_w  = (const float*)d_in[5];
  const float* comb_b  = (const float*)d_in[6];
  const float* mu_w    = (const float*)d_in[7];
  const float* mu_b    = (const float*)d_in[8];
  const float* sig_w   = (const float*)d_in[9];
  const float* sig_b   = (const float*)d_in[10];
  const float* em_w1   = (const float*)d_in[11];
  const float* em_b1   = (const float*)d_in[12];
  const float* em_w2   = (const float*)d_in[13];
  const float* em_b2   = (const float*)d_in[14];
  const float* em_w3   = (const float*)d_in[15];
  const float* em_b3   = (const float*)d_in[16];
  const float* g_w1    = (const float*)d_in[17];
  const float* g_b1    = (const float*)d_in[18];
  const float* g_w2    = (const float*)d_in[19];
  const float* g_b2    = (const float*)d_in[20];
  const float* h_w1    = (const float*)d_in[21];
  const float* h_b1    = (const float*)d_in[22];
  const float* h_w2    = (const float*)d_in[23];
  const float* h_b2    = (const float*)d_in[24];
  const float* mg_w    = (const float*)d_in[25];
  const float* mg_b    = (const float*)d_in[26];
  const float* sg_w    = (const float*)d_in[27];
  const float* sg_b    = (const float*)d_in[28];
  const float* eps_c   = (const float*)d_in[29];
  const float* eps_i   = (const float*)d_in[30];
  const float* eps_t   = (const float*)d_in[31];

  float* x_hat   = (float*)d_out;
  float* mus_inf = x_hat + 128 * 512 * 88;
  float* sig_inf = mus_inf + 128 * 512 * 100;
  float* mus_gen = sig_inf + 128 * 512 * 100;
  float* sig_gen = mus_gen + 128 * 512 * 100;

  // workspace carve (bf16 padded weight copies + staged x + z_hat)
  size_t off = 0;
  auto carve = [&](size_t bytes) {
    void* p = (char*)d_ws + off;
    off += (bytes + 255) & ~(size_t)255;
    return p;
  };
  u16* WIH  = (u16*)carve(400 * 96 * 2);
  u16* WHH  = (u16*)carve(400 * 416 * 2);
  u16* CW   = (u16*)carve(400 * 128 * 2);
  u16* MUW  = (u16*)carve(112 * 416 * 2);
  u16* SIW  = (u16*)carve(112 * 416 * 2);
  u16* EW1  = (u16*)carve(112 * 128 * 2);
  u16* EW2  = (u16*)carve(112 * 128 * 2);
  u16* EW3  = (u16*)carve(96 * 128 * 2);
  u16* GW1  = (u16*)carve(208 * 128 * 2);
  u16* GW2  = (u16*)carve(112 * 224 * 2);
  u16* HW1  = (u16*)carve(208 * 128 * 2);
  u16* HW2  = (u16*)carve(112 * 224 * 2);
  u16* MGW  = (u16*)carve(112 * 128 * 2);
  u16* SGW2 = (u16*)carve(112 * 128 * 2);
  u16* XBF  = (u16*)carve((size_t)512 * 128 * 96 * 2);
  u16* ZHAT = (u16*)carve((size_t)65536 * 128 * 2);

  auto cdiv = [](int a, int b) { return (a + b - 1) / b; };
  auto cvt = [&](const float* in, u16* out, int R, int C, int Rp, int Cp) {
    cvt_pad<<<cdiv(Rp * Cp, 256), 256, 0, stream>>>(in, out, R, C, Rp, Cp);
  };
  cvt(w_ih,   WIH, 400, 88, 400, 96);
  cvt(w_hh,   WHH, 400, 400, 400, 416);
  cvt(comb_w, CW,  400, 100, 400, 128);
  cvt(mu_w,   MUW, 100, 400, 112, 416);
  cvt(sig_w,  SIW, 100, 400, 112, 416);
  cvt(em_w1,  EW1, 100, 100, 112, 128);
  cvt(em_w2,  EW2, 100, 100, 112, 128);
  cvt(em_w3,  EW3, 88, 100, 96, 128);
  cvt(g_w1,   GW1, 200, 100, 208, 128);
  cvt(g_w2,   GW2, 100, 200, 112, 224);
  cvt(h_w1,   HW1, 200, 100, 208, 128);
  cvt(h_w2,   HW2, 100, 200, 112, 224);
  cvt(mg_w,   MGW, 100, 100, 112, 128);
  cvt(sg_w,   SGW2, 100, 100, 112, 128);

  cvt_x<<<cdiv(512 * 128 * 96, 256), 256, 0, stream>>>(x, XBF);
  zero_us<<<cdiv(65536 * 128, 256), 256, 0, stream>>>(ZHAT, 65536 * 128);

  dvae_scan<<<16, 256, 0, stream>>>(
      XBF, b_ih, b_hh, comb_b, mu_b, sig_b, g_b1, g_b2, h_b1, h_b2, mg_b, sg_b,
      eps_c, eps_i, eps_t,
      WIH, WHH, CW, MUW, SIW, GW1, GW2, HW1, HW2, MGW, SGW2,
      ZHAT, mus_inf, sig_inf, mus_gen, sig_gen);

  dvae_emission<<<4096, 128, 0, stream>>>(ZHAT, EW1, EW2, EW3, em_b1, em_b2, em_b3, x_hat);
}
